// LstmDecoder_1829656068528
// MI455X (gfx1250) — compile-verified, matrix operand order checked by reading, not run on hardware
//
#include <hip/hip_runtime.h>
#include <hip/hip_bf16.h>
#include <math.h>

#define B_   64
#define T_   512
#define IN_  256
#define OUT_ 256
#define H_   1024
#define G4_  4096
#define NWG  64

typedef __attribute__((ext_vector_type(16))) __bf16        v16bf;
typedef __attribute__((ext_vector_type(8)))  float         v8f;
typedef __attribute__((ext_vector_type(4)))  unsigned int  v4u;

__device__ __forceinline__ unsigned short f2bf(float f) {
  unsigned int u = __float_as_uint(f);
  u += 0x7FFFu + ((u >> 16) & 1u);          // round-to-nearest-even
  return (unsigned short)(u >> 16);
}

__device__ __forceinline__ float sigmoidf(float x) {
  return 1.0f / (1.0f + __expf(-x));
}
__device__ __forceinline__ float fast_tanh(float x) {
  float e = __expf(2.0f * x);
  return (e - 1.0f) / (e + 1.0f);
}

// Load one 16-lane-striped bf16 WMMA fragment (A: r0 = batch-row base,
// B: r0 = gate-row base of row-major [N,K] weight). Per the CDNA5 16-bit
// layout, lane L holds row r0+(L&15), K chunks [k0+o, k0+o+8) and
// [k0+o+16, k0+o+24) with o = (L>=16)*8  -> two 16-byte contiguous loads.
__device__ __forceinline__ v16bf load_frag16(const unsigned short* __restrict__ base,
                                             int r0, int stride, int k0) {
  const int lane = threadIdx.x & 31;
  const int r = r0 + (lane & 15);
  const int k = k0 + ((lane >> 4) << 3);
  const unsigned short* p = base + (size_t)r * (size_t)stride + k;
  union { v16bf v; v4u u[2]; } f;
  f.u[0] = *(const v4u*)(p);
  f.u[1] = *(const v4u*)(p + 16);
  return f.v;
}

__device__ __forceinline__ v8f wmma_bf16(v16bf a, v16bf b, v8f c) {
  return __builtin_amdgcn_wmma_f32_16x16x32_bf16(false, a, false, b, (short)0, c,
                                                 false, false);
}

// Monotonic-counter grid barrier (64 co-resident persistent WGs).
__device__ __forceinline__ void grid_barrier(unsigned int* bar, unsigned int epoch) {
  __syncthreads();
  if (threadIdx.x == 0) {
    __threadfence();
    __hip_atomic_fetch_add(bar, 1u, __ATOMIC_ACQ_REL, __HIP_MEMORY_SCOPE_AGENT);
    const unsigned int tgt = epoch * NWG;
    while (__hip_atomic_load(bar, __ATOMIC_ACQUIRE, __HIP_MEMORY_SCOPE_AGENT) < tgt) {
      __builtin_amdgcn_s_sleep(1);
    }
    __threadfence();
  }
  __syncthreads();
}

// Elementwise LSTM cell. Waves 4-7 hold (g,o) pre-activations and pass them
// through LDS to waves 0-3, which hold (i,f) and the persistent c state.
__device__ __forceinline__ void cell_update(v8f acc_a, v8f acc_b, int hi, int mt,
                                            int m0, int hcol0, int col, int lane,
                                            float bias_a, float bias_b, v8f& c,
                                            unsigned short* __restrict__ hdst,
                                            float (*lds)[2][8][32]) {
  if (hi == 1) {
#pragma unroll
    for (int r = 0; r < 8; ++r) {
      lds[mt][0][r][lane] = acc_a[r] + bias_a;  // g pre-act
      lds[mt][1][r][lane] = acc_b[r] + bias_b;  // o pre-act
    }
  }
  __syncthreads();
  if (hi == 0) {
#pragma unroll
    for (int r = 0; r < 8; ++r) {
      float ig = sigmoidf(acc_a[r] + bias_a);
      float fg = sigmoidf(acc_b[r] + bias_b);
      float gg = fast_tanh(lds[mt][0][r][lane]);
      float og = sigmoidf(lds[mt][1][r][lane]);
      float cn = fg * c[r] + ig * gg;
      c[r] = cn;
      float hn = og * fast_tanh(cn);
      const int row = m0 + r + ((lane >> 4) << 3);   // C/D accumulator row map
      hdst[(size_t)row * H_ + hcol0 + col] = f2bf(hn);
    }
  }
}

__global__ void k_f32_to_bf16(const float* __restrict__ s,
                              unsigned short* __restrict__ d, int n) {
  int i = blockIdx.x * blockDim.x + threadIdx.x;
  if (i < n) d[i] = f2bf(s[i]);
}

__global__ void k_init(const float* __restrict__ z,
                       unsigned short* __restrict__ h0a,
                       unsigned short* __restrict__ h1a,
                       unsigned int* __restrict__ bar) {
  int i = blockIdx.x * blockDim.x + threadIdx.x;
  if (i < B_ * H_) {
    unsigned short v = f2bf(z[i]);
    h0a[i] = v;
    h1a[i] = v;
  }
  if (i == 0) *bar = 0u;
}

__global__ __launch_bounds__(256)
void lstm_persistent(const unsigned short* __restrict__ Wih0,
                     const unsigned short* __restrict__ Whh0,
                     const unsigned short* __restrict__ Wih1,
                     const unsigned short* __restrict__ Whh1,
                     const unsigned short* __restrict__ Wlin,
                     const unsigned short* __restrict__ Xbf,
                     unsigned short* __restrict__ h0p, unsigned short* __restrict__ h0q,
                     unsigned short* __restrict__ h1p, unsigned short* __restrict__ h1q,
                     const float* __restrict__ bih0, const float* __restrict__ bhh0,
                     const float* __restrict__ bih1, const float* __restrict__ bhh1,
                     const float* __restrict__ blin,
                     unsigned int* __restrict__ bar,
                     float* __restrict__ out) {
  const int wg    = blockIdx.x;          // 0..63 : 16-wide slice of H
  const int wave  = threadIdx.x >> 5;    // 0..7
  const int lane  = threadIdx.x & 31;
  const int mt    = wave & 3;            // batch m-tile (16 rows)
  const int hi    = wave >> 2;           // 0 -> gates (i,f), 1 -> gates (g,o)
  const int m0    = mt << 4;
  const int hcol0 = wg << 4;
  const int col   = lane & 15;

  __shared__ float lds_go[4][2][8][32];  // 8 KB: g,o pre-acts per m-tile

  v8f c0 = {};
  v8f c1 = {};
  unsigned int epoch = 0;

  const int ga = hi * 2, gb = ga + 1;    // gate group indices for this wave
  const int na = ga * H_ + hcol0;        // gate rows in [4H,K] weight
  const int nb = gb * H_ + hcol0;

  for (int t = 0; t < T_; ++t) {
    const int pr = t & 1;
    // Ping-pong buffers selected as SSA values (no stack array!) so the
    // address-space inference keeps these as global_load/global_store,
    // not FLAT (which would also tick DScnt and serialize vs. LDS).
    const unsigned short* h0r = pr ? h0q : h0p;
    unsigned short*       h0w = pr ? h0p : h0q;
    const unsigned short* h1r = pr ? h1q : h1p;
    unsigned short*       h1w = pr ? h1p : h1q;

    // ---------------- layer 0: gates0 = x_t·Wih0^T + h0·Whh0^T ----------------
    v8f acc_a = {};
    v8f acc_b = {};
    {
#pragma unroll 4
      for (int k = 0; k < H_; k += 32) {
        v16bf a  = load_frag16(h0r, m0, H_, k);
        v16bf ba = load_frag16(Whh0, na, H_, k);
        v16bf bb = load_frag16(Whh0, nb, H_, k);
        acc_a = wmma_bf16(a, ba, acc_a);
        acc_b = wmma_bf16(a, bb, acc_b);
      }
      if (t > 0) {  // step 0 input is zeros -> skip
        const unsigned short* xbase = Xbf + (size_t)(t - 1) * IN_;  // x[:, t-1, :]
#pragma unroll 4
        for (int k = 0; k < IN_; k += 32) {
          v16bf a  = load_frag16(xbase, m0, T_ * IN_, k);
          v16bf ba = load_frag16(Wih0, na, IN_, k);
          v16bf bb = load_frag16(Wih0, nb, IN_, k);
          acc_a = wmma_bf16(a, ba, acc_a);
          acc_b = wmma_bf16(a, bb, acc_b);
        }
      }
    }
    cell_update(acc_a, acc_b, hi, mt, m0, hcol0, col, lane,
                bih0[na + col] + bhh0[na + col],
                bih0[nb + col] + bhh0[nb + col],
                c0, h0w, lds_go);
    grid_barrier(bar, ++epoch);   // all h0 slices visible

    // ---------------- layer 1: gates1 = h0·Wih1^T + h1·Whh1^T ----------------
    acc_a = (v8f){};
    acc_b = (v8f){};
    {
#pragma unroll 4
      for (int k = 0; k < H_; k += 32) {
        v16bf a  = load_frag16(h0w, m0, H_, k);
        v16bf ba = load_frag16(Wih1, na, H_, k);
        v16bf bb = load_frag16(Wih1, nb, H_, k);
        acc_a = wmma_bf16(a, ba, acc_a);
        acc_b = wmma_bf16(a, bb, acc_b);
      }
#pragma unroll 4
      for (int k = 0; k < H_; k += 32) {
        v16bf a  = load_frag16(h1r, m0, H_, k);
        v16bf ba = load_frag16(Whh1, na, H_, k);
        v16bf bb = load_frag16(Whh1, nb, H_, k);
        acc_a = wmma_bf16(a, ba, acc_a);
        acc_b = wmma_bf16(a, bb, acc_b);
      }
    }
    cell_update(acc_a, acc_b, hi, mt, m0, hcol0, col, lane,
                bih1[na + col] + bhh1[na + col],
                bih1[nb + col] + bhh1[nb + col],
                c1, h1w, lds_go);
    grid_barrier(bar, ++epoch);   // all h1 slices visible

    // ---------------- head: out_t = h1·Wlin^T + blin (WGs 0-15) ----------------
    if (wg < 16 && hi == 0) {     // wave-uniform branch: EXEC all-1 inside
      v8f acc = {};
      const int n0 = wg << 4;
#pragma unroll 4
      for (int k = 0; k < H_; k += 32) {
        v16bf a = load_frag16(h1w, m0, H_, k);
        v16bf b = load_frag16(Wlin, n0, H_, k);
        acc = wmma_bf16(a, b, acc);
      }
      const float bl = blin[n0 + col];
#pragma unroll
      for (int r = 0; r < 8; ++r) {
        const int row = m0 + r + ((lane >> 4) << 3);
        out[(size_t)row * (T_ * OUT_) + (size_t)t * OUT_ + n0 + col] = acc[r] + bl;
      }
    }
  }
}

extern "C" void kernel_launch(void* const* d_in, const int* in_sizes, int n_in,
                              void* d_out, int out_size, void* d_ws, size_t ws_size,
                              hipStream_t stream) {
  const float* z    = (const float*)d_in[0];
  const float* x    = (const float*)d_in[1];
  const float* Wih0 = (const float*)d_in[2];
  const float* Whh0 = (const float*)d_in[3];
  const float* bih0 = (const float*)d_in[4];
  const float* bhh0 = (const float*)d_in[5];
  const float* Wih1 = (const float*)d_in[6];
  const float* Whh1 = (const float*)d_in[7];
  const float* bih1 = (const float*)d_in[8];
  const float* bhh1 = (const float*)d_in[9];
  const float* Wlin = (const float*)d_in[10];
  const float* blin = (const float*)d_in[11];

  // Workspace layout (bf16 copies + ping-pong h buffers + barrier), ~45 MB.
  unsigned short* ws = (unsigned short*)d_ws;
  size_t off = 0;
  auto take = [&](size_t n) { unsigned short* p = ws + off; off += n; return p; };
  unsigned short* wih0 = take((size_t)G4_ * IN_);
  unsigned short* whh0 = take((size_t)G4_ * H_);
  unsigned short* wih1 = take((size_t)G4_ * H_);
  unsigned short* whh1 = take((size_t)G4_ * H_);
  unsigned short* wlin = take((size_t)OUT_ * H_);
  unsigned short* xbf  = take((size_t)B_ * T_ * IN_);
  unsigned short* h0a  = take((size_t)B_ * H_);
  unsigned short* h0b  = take((size_t)B_ * H_);
  unsigned short* h1a  = take((size_t)B_ * H_);
  unsigned short* h1b  = take((size_t)B_ * H_);
  unsigned int*   bar  = (unsigned int*)(ws + off);

  auto cvt = [&](const float* s, unsigned short* d, int n) {
    k_f32_to_bf16<<<(n + 255) / 256, 256, 0, stream>>>(s, d, n);
  };
  cvt(Wih0, wih0, G4_ * IN_);
  cvt(Whh0, whh0, G4_ * H_);
  cvt(Wih1, wih1, G4_ * H_);
  cvt(Whh1, whh1, G4_ * H_);
  cvt(Wlin, wlin, OUT_ * H_);
  cvt(x, xbf, B_ * T_ * IN_);
  k_init<<<(B_ * H_ + 255) / 256, 256, 0, stream>>>(z, h0a, h1a, bar);

  lstm_persistent<<<NWG, 256, 0, stream>>>(wih0, whh0, wih1, whh1, wlin, xbf,
                                           h0a, h0b, h1a, h1b,
                                           bih0, bhh0, bih1, bhh1, blin,
                                           bar, (float*)d_out);
}
